// SelfAttention_56839597195376
// MI455X (gfx1250) — matrix operand order repaired and checked by reading, not compiled, not run on hardware
//
#include <hip/hip_runtime.h>
#include <hip/hip_bf16.h>

typedef __attribute__((ext_vector_type(16))) _Float16 v16h;
typedef __attribute__((ext_vector_type(8)))  _Float16 v8h;
typedef __attribute__((ext_vector_type(8)))  float    v8f;
typedef __attribute__((ext_vector_type(4)))  int      v4i;

#define WMMA_F16(a, b, c) \
  __builtin_amdgcn_wmma_f32_16x16x32_f16(false, (a), false, (b), (short)0, (c), false, false)

static constexpr int Bc = 4, Lc = 2048, Cc = 1024, Hc = 16, Dc = 64;

// ---------------- gfx1250 async global->LDS path (guarded; sync fallback) ----------------
#if defined(__has_builtin)
#if __has_builtin(__builtin_amdgcn_global_load_async_to_lds_b128)
#define HAVE_ASYNC_LDS 1
#endif
#if __has_builtin(__builtin_amdgcn_s_wait_asynccnt)
#define HAVE_WAIT_ASYNC 1
#endif
#endif

#define AS1 __attribute__((address_space(1)))
#define AS3 __attribute__((address_space(3)))

__device__ __forceinline__ void async_copy_b128(const _Float16* g, _Float16* l) {
#if defined(HAVE_ASYNC_LDS)
  __builtin_amdgcn_global_load_async_to_lds_b128(
      (AS1 v4i*)(unsigned long long)g, (AS3 v4i*)l, 0, 0);
#else
  *(v8h*)l = *(const v8h*)g;
#endif
}

__device__ __forceinline__ void wait_async_all() {
#if defined(HAVE_ASYNC_LDS)
#if defined(HAVE_WAIT_ASYNC)
  __builtin_amdgcn_s_wait_asynccnt(0);
#else
  asm volatile("s_wait_asynccnt 0x0" ::: "memory");
#endif
#endif
}

// ---------------- fragment loaders (CDNA5 wave32 WMMA layouts) ----------------
// A (16x32 f16): m = lane&15, k(e) = e + (e&8) + ((lane&16)>>1)
__device__ __forceinline__ v16h load_a_frag(const _Float16* p, int strideH) {
  const int lane = threadIdx.x & 31;
  const int m   = lane & 15;
  const int khi = (lane & 16) >> 1;           // 0 or 8
  const _Float16* row = p + m * strideH + khi;
  v8h lo = *(const v8h*)(row);                // k = khi + 0..7
  v8h hi = *(const v8h*)(row + 16);           // k = khi + 16..23
  v16h a;
#pragma unroll
  for (int i = 0; i < 8; ++i) { a[i] = lo[i]; a[i + 8] = hi[i]; }
  return a;
}

// B (32x16 f16) from LDS laid out [n][k]: n = lane&15, k(e) = e + (e&8) + ((lane&16)>>1)
__device__ __forceinline__ v16h load_b_frag(const _Float16* p, int strideH) {
  const int lane = threadIdx.x & 31;
  const int n  = lane & 15;
  const int kb = (lane & 16) >> 1;            // 0 or 8
  const _Float16* row = p + n * strideH + kb;
  v8h lo = *(const v8h*)(row);
  v8h hi = *(const v8h*)(row + 16);
  v16h b;
#pragma unroll
  for (int i = 0; i < 8; ++i) { b[i] = lo[i]; b[i + 8] = hi[i]; }
  return b;
}

// ---------------- f32 -> f16 convert ----------------
__global__ void cvt_f32_f16(const float* __restrict__ in, _Float16* __restrict__ out, int n) {
  int i = blockIdx.x * 256 + threadIdx.x;
  if (i < n) out[i] = (_Float16)in[i];
}

// ---------------- NT GEMM: out[m,n] = sum_k A[m,k]*B[n,k]  (128x128 tile) ----------------
// MODE 0: qkv epilogue (bias + scatter to [3][B][H][L][64] f16)
// MODE 1: proj epilogue (+ b_proj, f32 out row-major [M,N])
template <int MODE>
__global__ __launch_bounds__(256) void gemm_nt_kernel(
    const _Float16* __restrict__ A, const _Float16* __restrict__ Bm, int K, int N,
    const float* __restrict__ qb, const float* __restrict__ vb,
    _Float16* __restrict__ qkv_out,
    const float* __restrict__ bproj, float* __restrict__ fout) {
  __shared__ __align__(16) _Float16 a_lds[128][40];
  __shared__ __align__(16) _Float16 b_lds[128][40];
  const int tid = threadIdx.x;
  const int w = tid >> 5, lane = tid & 31;
  const int wm = w >> 1, wn = w & 1;          // 4 x 2 wave grid; wave = 32 rows x 64 cols
  const int n0 = blockIdx.x * 128, m0 = blockIdx.y * 128;

  v8f c[2][4] = {};
  const int arow = tid >> 1, acol = (tid & 1) * 16;   // 128 x 32 cooperative copy

  for (int k0 = 0; k0 < K; k0 += 32) {
    __syncthreads();
    const _Float16* asrc = A + (size_t)(m0 + arow) * K + k0 + acol;
    const _Float16* bsrc = Bm + (size_t)(n0 + arow) * K + k0 + acol;
    *(v8h*)&a_lds[arow][acol]     = *(const v8h*)asrc;
    *(v8h*)&a_lds[arow][acol + 8] = *(const v8h*)(asrc + 8);
    *(v8h*)&b_lds[arow][acol]     = *(const v8h*)bsrc;
    *(v8h*)&b_lds[arow][acol + 8] = *(const v8h*)(bsrc + 8);
    __syncthreads();
    v16h a0 = load_a_frag(&a_lds[wm * 32][0], 40);
    v16h a1 = load_a_frag(&a_lds[wm * 32 + 16][0], 40);
#pragma unroll
    for (int j = 0; j < 4; ++j) {
      v16h bj = load_b_frag(&b_lds[wn * 64 + j * 16][0], 40);
      c[0][j] = WMMA_F16(a0, bj, c[0][j]);
      c[1][j] = WMMA_F16(a1, bj, c[1][j]);
    }
  }

  const int nlane = lane & 15, rowbase = (lane & 16) >> 1;
#pragma unroll
  for (int i = 0; i < 2; ++i) {
#pragma unroll
    for (int j = 0; j < 4; ++j) {
      const int n = n0 + wn * 64 + j * 16 + nlane;
#pragma unroll
      for (int r = 0; r < 8; ++r) {
        const int m = m0 + wm * 32 + i * 16 + rowbase + r;
        float val = c[i][j][r];
        if (MODE == 0) {
          float bias = (n < Cc) ? qb[n] : ((n >= 2 * Cc) ? vb[n - 2 * Cc] : 0.0f);
          val += bias;
          const int which = n >> 10, cidx = n & (Cc - 1);
          const int hh = cidx >> 6, dd = cidx & 63;
          const int bb = m >> 11, ll = m & (Lc - 1);
          qkv_out[((((size_t)which * Bc + bb) * Hc + hh) * Lc + ll) * Dc + dd] = (_Float16)val;
        } else {
          fout[(size_t)m * N + n] = val + bproj[n];
        }
      }
    }
  }
}

// ---------------- L2 normalize q (x scale) and k : one wave per 64-wide row ----------------
__global__ __launch_bounds__(256) void normalize_qk(_Float16* __restrict__ qkv,
                                                    const float* __restrict__ smlog,
                                                    int rowsQ) {
  const int w = threadIdx.x >> 5, lane = threadIdx.x & 31;
  const int rid = blockIdx.x * 8 + w;          // [0, 2*rowsQ): q rows then k rows
  const bool isq = rid < rowsQ;
  const int r = isq ? rid : rid - rowsQ;
  _Float16* base = qkv + (size_t)rid * Dc;
  float x0 = (float)base[lane * 2];
  float x1 = (float)base[lane * 2 + 1];
  float s = x0 * x0 + x1 * x1;
#pragma unroll
  for (int msk = 16; msk >= 1; msk >>= 1) s += __shfl_xor(s, msk, 32);
  float nrm = fmaxf(sqrtf(s), 1e-12f);
  float sc = 1.0f;
  if (isq) {
    const int h = (r / Lc) & (Hc - 1);
    sc = __expf(fminf(smlog[h], 4.6051702f));  // log(100)
  }
  const float inv = sc / nrm;
  base[lane * 2]     = (_Float16)(x0 * inv);
  base[lane * 2 + 1] = (_Float16)(x1 * inv);
}

// ---------------- fused flash attention ----------------
// block = (b, h, 128 q rows); each wave owns 16 q rows end-to-end (no cross-wave softmax state).
// Double-buffered K (async global->LDS when available) and V (register-staged transpose).
// Row-sum of P computed as an extra WMMA against an all-ones fragment.
__global__ __launch_bounds__(256) void attn_kernel(const _Float16* __restrict__ qbuf,
                                                   const _Float16* __restrict__ kbuf,
                                                   const _Float16* __restrict__ vbuf,
                                                   const float* __restrict__ bias,
                                                   _Float16* __restrict__ outh) {
  __shared__ __align__(16) _Float16 k_lds[2][32][72];     // [buf][kv][d]
  __shared__ __align__(16) _Float16 v_t[2][64][40];       // [buf][d][kv] (transposed)
  __shared__ __align__(16) _Float16 p_lds[8][16][40];     // per-wave P tile
  const int tid = threadIdx.x, w = tid >> 5, lane = tid & 31;
  const int b = blockIdx.z, h = blockIdx.y, q0 = blockIdx.x * 128;
  const int nlane = lane & 15, rowbase = (lane & 16) >> 1;
  const int qrow0 = q0 + w * 16;
  const size_t headoff = (size_t)(b * Hc + h) * Lc * Dc;
  constexpr int NIT = Lc / 32;                            // 64 kv chunks

  // q fragments (16 rows x 64 d), resident for the whole kv loop
  v16h qa0 = load_a_frag(qbuf + headoff + (size_t)qrow0 * Dc, Dc);       // d 0..31
  v16h qa1 = load_a_frag(qbuf + headoff + (size_t)qrow0 * Dc + 32, Dc);  // d 32..63

  v16h ones;
#pragma unroll
  for (int i = 0; i < 16; ++i) ones[i] = (_Float16)1.0f;

  float m_i[8];
  v8f o[4] = {};
  v8f lacc = {};                                          // row sums of P (every column equal)
#pragma unroll
  for (int r = 0; r < 8; ++r) m_i[r] = -1e30f;

  const float* bias_base = bias + ((size_t)h * Lc + (qrow0 + rowbase)) * Lc + nlane;
  const int ldrow = tid >> 3, ldcol = (tid & 7) * 8;      // cooperative 32x64 tile copy
  const _Float16* kg = kbuf + headoff + (size_t)ldrow * Dc + ldcol;
  const _Float16* vg = vbuf + headoff + (size_t)ldrow * Dc + ldcol;

  // ---- stage chunk 0 ----
  async_copy_b128(kg, &k_lds[0][ldrow][ldcol]);
  {
    v8h vv = *(const v8h*)vg;
#pragma unroll
    for (int t = 0; t < 8; ++t) v_t[0][ldcol + t][ldrow] = vv[t];
  }

  for (int it = 0; it < NIT; ++it) {
    const int kv0 = it * 32;
    const int cur = it & 1, nxt = cur ^ 1;
    wait_async_all();
    __syncthreads();                                      // chunk `it` visible; prev buf free

    // ---- start fetch of chunk it+1 ----
    v8h vreg;
#if !defined(HAVE_ASYNC_LDS)
    v8h kreg;
#endif
    if (it + 1 < NIT) {
      const size_t off = (size_t)(kv0 + 32) * Dc;
#if defined(HAVE_ASYNC_LDS)
      async_copy_b128(kg + off, &k_lds[nxt][ldrow][ldcol]);
#else
      kreg = *(const v8h*)(kg + off);
#endif
      vreg = *(const v8h*)(vg + off);
    }
    __builtin_prefetch(bias_base + kv0 + 32, 0, 3);       // next chunk of the bias stream

    // ---- S = q @ k^T (16x32), K-dim 64 -> 2 WMMAs per 16-col subtile ----
    v8f s0 = {}, s1 = {};
    s0 = WMMA_F16(qa0, load_b_frag(&k_lds[cur][0][0], 72), s0);
    s0 = WMMA_F16(qa1, load_b_frag(&k_lds[cur][0][32], 72), s0);
    s1 = WMMA_F16(qa0, load_b_frag(&k_lds[cur][16][0], 72), s1);
    s1 = WMMA_F16(qa1, load_b_frag(&k_lds[cur][16][32], 72), s1);

    // ---- online softmax: row max via lane shuffles, row sum via ones-WMMA ----
    const float* bp = bias_base + kv0;
#pragma unroll
    for (int r = 0; r < 8; ++r) {
      float sv0 = s0[r] + bp[(size_t)r * Lc];
      float sv1 = s1[r] + bp[(size_t)r * Lc + 16];
      float t = fmaxf(sv0, sv1);
#pragma unroll
      for (int msk = 8; msk >= 1; msk >>= 1) t = fmaxf(t, __shfl_xor(t, msk, 32));
      const float mn = fmaxf(m_i[r], t);
      const float corr = __expf(m_i[r] - mn);
      m_i[r] = mn;
      const float p0 = __expf(sv0 - mn);
      const float p1 = __expf(sv1 - mn);
      o[0][r] *= corr; o[1][r] *= corr; o[2][r] *= corr; o[3][r] *= corr;
      lacc[r] *= corr;
      p_lds[w][rowbase + r][nlane]      = (_Float16)p0;
      p_lds[w][rowbase + r][16 + nlane] = (_Float16)p1;
    }

    // ---- O += P @ V ; l += P @ ones ----
    v16h pa = load_a_frag(&p_lds[w][0][0], 40);
    o[0] = WMMA_F16(pa, load_b_frag(&v_t[cur][0][0], 40), o[0]);
    o[1] = WMMA_F16(pa, load_b_frag(&v_t[cur][16][0], 40), o[1]);
    o[2] = WMMA_F16(pa, load_b_frag(&v_t[cur][32][0], 40), o[2]);
    o[3] = WMMA_F16(pa, load_b_frag(&v_t[cur][48][0], 40), o[3]);
    lacc = WMMA_F16(pa, ones, lacc);

    // ---- finish staging chunk it+1 ----
    if (it + 1 < NIT) {
#if !defined(HAVE_ASYNC_LDS)
      *(v8h*)&k_lds[nxt][ldrow][ldcol] = kreg;
#endif
#pragma unroll
      for (int t = 0; t < 8; ++t) v_t[nxt][ldcol + t][ldrow] = vreg[t];
    }
  }

#pragma unroll
  for (int r = 0; r < 8; ++r) {
    const float inv = 1.0f / lacc[r];
    const size_t orow = (size_t)(b * Lc + qrow0 + rowbase + r) * Cc + h * Dc + nlane;
    outh[orow]      = (_Float16)(o[0][r] * inv);
    outh[orow + 16] = (_Float16)(o[1][r] * inv);
    outh[orow + 32] = (_Float16)(o[2][r] * inv);
    outh[orow + 48] = (_Float16)(o[3][r] * inv);
  }
}

// ---------------- host launcher ----------------
extern "C" void kernel_launch(void* const* d_in, const int* in_sizes, int n_in,
                              void* d_out, int out_size, void* d_ws, size_t ws_size,
                              hipStream_t stream) {
  const float* x      = (const float*)d_in[0];
  const float* abias  = (const float*)d_in[1];
  const float* W_qkv  = (const float*)d_in[2];
  const float* q_bias = (const float*)d_in[3];
  const float* v_bias = (const float*)d_in[4];
  const float* smlog  = (const float*)d_in[5];
  const float* W_proj = (const float*)d_in[6];
  const float* b_proj = (const float*)d_in[7];
  float* out = (float*)d_out;

  const int NX    = Bc * Lc * Cc;        // 8388608
  const int NWQKV = 3 * Cc * Cc;         // 3145728
  const int NWPRJ = Cc * Cc;             // 1048576
  const int NQKV3 = 3 * Bc * Hc * Lc * Dc;

  _Float16* xh     = (_Float16*)d_ws;
  _Float16* wqkvh  = xh + NX;
  _Float16* wprojh = wqkvh + NWQKV;
  _Float16* qkvh   = wprojh + NWPRJ;     // [3][B][H][L][64]
  _Float16* attnh  = qkvh + NQKV3;       // [B*L, C]

  cvt_f32_f16<<<(NX + 255) / 256, 256, 0, stream>>>(x, xh, NX);
  cvt_f32_f16<<<(NWQKV + 255) / 256, 256, 0, stream>>>(W_qkv, wqkvh, NWQKV);
  cvt_f32_f16<<<(NWPRJ + 255) / 256, 256, 0, stream>>>(W_proj, wprojh, NWPRJ);

  {
    dim3 grid(3 * Cc / 128, (Bc * Lc) / 128);
    gemm_nt_kernel<0><<<grid, 256, 0, stream>>>(xh, wqkvh, Cc, 3 * Cc,
                                                q_bias, v_bias, qkvh, nullptr, nullptr);
  }
  {
    const int rowsQ = Bc * Hc * Lc;      // 131072
    normalize_qk<<<(2 * rowsQ) / 8, 256, 0, stream>>>(qkvh, smlog, rowsQ);
  }
  {
    const size_t perTensor = (size_t)Bc * Hc * Lc * Dc;
    dim3 grid(Lc / 128, Hc, Bc);
    attn_kernel<<<grid, 256, 0, stream>>>(qkvh, qkvh + perTensor, qkvh + 2 * perTensor,
                                          abias, attnh);
  }
  {
    dim3 grid(Cc / 128, (Bc * Lc) / 128);
    gemm_nt_kernel<1><<<grid, 256, 0, stream>>>(attnh, wprojh, Cc, Cc,
                                                nullptr, nullptr, nullptr, b_proj, out);
  }
}
